// GTModel_9062380995149
// MI455X (gfx1250) — matrix operand order repaired and checked by reading, not compiled, not run on hardware
//
#include <hip/hip_runtime.h>
#include <hip/hip_bf16.h>
#include <math.h>

#define IN_F   128
#define HID_F  256
#define NLAYER 8
#define HEADS  8
#define OUT_F  40
#define DH     32

typedef __attribute__((ext_vector_type(16))) __bf16    v16bf;
typedef __attribute__((ext_vector_type(8)))  float     v8f;
typedef __attribute__((ext_vector_type(8)))  unsigned  v8u;
typedef __attribute__((ext_vector_type(4)))  unsigned  v4u;
typedef __attribute__((ext_vector_type(8)))  int       v8i;
typedef __attribute__((ext_vector_type(4)))  int       v4i;

#if __has_builtin(__builtin_amdgcn_tensor_load_to_lds)
#define HAVE_TDM 1
#else
#define HAVE_TDM 0
#endif

__device__ __forceinline__ __bf16 f2bf(float f) {
    unsigned u = __builtin_bit_cast(unsigned, f);
    unsigned r = (u + 0x7FFFu + ((u >> 16) & 1u)) >> 16;
    unsigned short s = (unsigned short)r;
    return __builtin_bit_cast(__bf16, s);
}

// unpack a packed pair of bf16 (low, high) to two floats
__device__ __forceinline__ float2 bfpair(unsigned u) {
    float lo = __builtin_bit_cast(float, u << 16);
    float hi = __builtin_bit_cast(float, u & 0xFFFF0000u);
    return make_float2(lo, hi);
}

// monotone float <-> uint encoding for atomicMax on floats
__device__ __forceinline__ unsigned fenc(float f) {
    unsigned u = __builtin_bit_cast(unsigned, f);
    return (u & 0x80000000u) ? ~u : (u | 0x80000000u);
}
__device__ __forceinline__ float fdec(unsigned u) {
    unsigned b = (u & 0x80000000u) ? (u & 0x7FFFFFFFu) : ~u;
    return __builtin_bit_cast(float, b);
}

// ---------------------------------------------------------------------------
// GEMM: C[n,M] = scale * (A[n,K] @ W[K,M] + bias[M]), A bf16 row-major,
// W pre-transposed+converted to Wt[M,K] bf16.
//  - A fragments loaded straight from global (L2-resident), ISA 16-bit layout
//  - B tile staged to LDS by the Tensor Data Mover (tensor_load_to_lds,
//    TENSORcnt) when available, else global_load_async_to_lds_b128 (ASYNCcnt).
//    Double buffered so the DMA overlaps the WMMAs.
// Block: 256 threads (8 waves); tile 128 rows x 64 cols; BK=32.
// ---------------------------------------------------------------------------
#define BM 128
#define BN 64
#define BK 32
#define LDB 40   // LDS row stride (bf16): 80B keeps 16B align, no bank conflicts

#if HAVE_TDM
// Issue one TDM descriptor copying a BN x BK tile of Wt[M,K] (bf16) into LDS.
// D# per CDNA5 ISA ch.8: 2-D tile, data_size=2B, HW OOB zero-fill beyond M,
// LDS padding 4 DWORDs every 16 DWORDs -> 80B row stride (== LDB elements).
__device__ __forceinline__ void stageB_tdm(const __bf16* Wt, unsigned ldsAddr,
                                           int colBase, int t, int K, int M)
{
    unsigned long long ga =
        (unsigned long long)(uintptr_t)(Wt + (size_t)colBase * K + (size_t)t * BK);
    v4u g0;
    g0[0] = 1u;                                        // count=1, no gather
    g0[1] = ldsAddr;                                   // lds_addr (bytes)
    g0[2] = (unsigned)(ga & 0xFFFFFFFFu);              // global_addr[31:0]
    g0[3] = (unsigned)((ga >> 32) & 0x01FFFFFFu)       // global_addr[56:32]
          | (2u << 30);                                // type=2 ("image")
    v8i g1;
    g1[0] = (int)((1u << 16)        // data_size = 2 bytes
                | (1u << 20)        // pad_enable
                | (3u << 22)        // pad_interval: 16 DWORDs (64B)
                | (3u << 25));      // pad_amount:   4 DWORDs (16B)
    g1[1] = (int)(((unsigned)K & 0xFFFFu) << 16);                  // dim0[15:0]
    g1[2] = (int)(((unsigned)K >> 16) | (((unsigned)M & 0xFFFFu) << 16)); // dim0[31:16], dim1[15:0]
    g1[3] = (int)(((unsigned)M >> 16) | ((unsigned)BK << 16));     // dim1[31:16], tile_dim0
    g1[4] = BN;                                                    // tile_dim1 (tile_dim2=0)
    g1[5] = K;                                                     // dim0_stride[31:0]
    g1[6] = 0;                                                     // dim0_stride hi, dim1_stride lo
    g1[7] = 0;
    v4i z4 = {0, 0, 0, 0};
    v8i z8 = {0, 0, 0, 0, 0, 0, 0, 0};
    __builtin_amdgcn_tensor_load_to_lds(g0, g1, z4, z4, z8, 0);
}
#endif

__global__ __launch_bounds__(256)
void gemm_wmma(const __bf16* __restrict__ A, const __bf16* __restrict__ Wt,
               const float* __restrict__ bias, void* __restrict__ Cout,
               int n, int K, int M, float scale, int out_bf16)
{
    __shared__ __align__(16) __bf16 Bs[2][BN * LDB];

    const int tid  = threadIdx.x;
    const int lane = tid & 31;
    const int wid  = tid >> 5;
    const int wrow = wid >> 1;    // 0..3 : 32-row band
    const int wcol = wid & 1;     // 0..1 : 32-col band
    const int half = lane >> 4;
    const int l16  = lane & 15;

    const int rowBase = blockIdx.y * BM;
    const int colBase = blockIdx.x * BN;

    const int ntiles = K / BK;

#if HAVE_TDM
    auto stageB = [&](int t, int buf) {
        if (wid == 0)   // one TDM issue per tile (EXEC ignored by TDM)
            stageB_tdm(Wt, (unsigned)(uintptr_t)&Bs[buf][0], colBase, t, K, M);
    };
#else
    // fallback: per-lane async copies (ASYNCcnt)
    const int brow = tid >> 2;        // 0..63
    const int bseg = (tid & 3) * 8;   // bf16 offset in k
    auto stageB = [&](int t, int buf) {
        int col = colBase + brow;
        if (col < M) {
            unsigned lds = (unsigned)(uintptr_t)&Bs[buf][brow * LDB + bseg];
            const void* g = (const void*)(Wt + (size_t)col * K + t * BK + bseg);
            asm volatile("global_load_async_to_lds_b128 %0, %1, off"
                         :: "v"(lds), "v"(g) : "memory");
        } else {
            uint4 z = {0u, 0u, 0u, 0u};
            *(uint4*)&Bs[buf][brow * LDB + bseg] = z;
        }
    };
#endif

    auto waitStage = [&]() {
#if HAVE_TDM
        __builtin_amdgcn_s_wait_tensorcnt(0);
#else
        asm volatile("s_wait_asynccnt 0x0" ::: "memory");
#endif
    };

    v8f acc[2][2] = {};

    stageB(0, 0);

    for (int t = 0; t < ntiles; ++t) {
        waitStage();
        __syncthreads();
        if (t + 1 < ntiles) stageB(t + 1, (t + 1) & 1);

        const int buf = t & 1;

        // ---- A fragments direct from global (16-bit A 16x32 lane layout) ----
        v16bf afrag[2];
#pragma unroll
        for (int i = 0; i < 2; ++i) {
            int gr = rowBase + wrow * 32 + i * 16 + l16;   // OOB rows read slack, masked at store
            const __bf16* ap = A + (size_t)gr * K + t * BK + (half ? 8 : 0);
            uint4 u0 = *(const uint4*)ap;          // half0: K0-7,  half1: K8-15
            uint4 u1 = *(const uint4*)(ap + 16);   // half0: K16-23, half1: K24-31
            v8u tv;
            tv[0] = u0.x; tv[1] = u0.y; tv[2] = u0.z; tv[3] = u0.w;
            tv[4] = u1.x; tv[5] = u1.y; tv[6] = u1.z; tv[7] = u1.w;
            afrag[i] = __builtin_bit_cast(v16bf, tv);
        }
        // ---- B fragments from LDS (lanes 0-15: K=0-7,16-23, lanes 16-31: K=8-15,24-31) ----
        v16bf bfrag[2];
#pragma unroll
        for (int j = 0; j < 2; ++j) {
            int cc = wcol * 32 + j * 16 + l16;
            const __bf16* bp = &Bs[buf][cc * LDB + half * 8];
            uint4 u0 = *(const uint4*)bp;
            uint4 u1 = *(const uint4*)(bp + 16);
            v8u tv;
            tv[0] = u0.x; tv[1] = u0.y; tv[2] = u0.z; tv[3] = u0.w;
            tv[4] = u1.x; tv[5] = u1.y; tv[6] = u1.z; tv[7] = u1.w;
            bfrag[j] = __builtin_bit_cast(v16bf, tv);
        }

#pragma unroll
        for (int i = 0; i < 2; ++i)
#pragma unroll
            for (int j = 0; j < 2; ++j)
                acc[i][j] = __builtin_amdgcn_wmma_f32_16x16x32_bf16(
                    false, afrag[i], false, bfrag[j],
                    (short)0, acc[i][j], false, false);
    }

    // ---- epilogue ----
    if (out_bf16) {
        __bf16* C = (__bf16*)Cout;
#pragma unroll
        for (int i = 0; i < 2; ++i)
#pragma unroll
            for (int j = 0; j < 2; ++j) {
                int col = colBase + wcol * 32 + j * 16 + l16;
                float b = (col < M) ? bias[col] : 0.f;
#pragma unroll
                for (int v = 0; v < 8; ++v) {
                    int row = rowBase + wrow * 32 + i * 16 + v + (half ? 8 : 0);
                    if (row < n && col < M)
                        C[(size_t)row * M + col] = f2bf((acc[i][j][v] + b) * scale);
                }
            }
    } else {
        float* C = (float*)Cout;
#pragma unroll
        for (int i = 0; i < 2; ++i)
#pragma unroll
            for (int j = 0; j < 2; ++j) {
                int col = colBase + wcol * 32 + j * 16 + l16;
                float b = (col < M) ? bias[col] : 0.f;
#pragma unroll
                for (int v = 0; v < 8; ++v) {
                    int row = rowBase + wrow * 32 + i * 16 + v + (half ? 8 : 0);
                    if (row < n && col < M)
                        C[(size_t)row * M + col] = (acc[i][j][v] + b) * scale;
                }
            }
    }
}

// ---------------------------------------------------------------------------
// conversion helpers
// ---------------------------------------------------------------------------
__global__ __launch_bounds__(256)
void conv_bf16(const float* __restrict__ in, __bf16* __restrict__ out, int total)
{
    int idx = blockIdx.x * blockDim.x + threadIdx.x;
    if (idx < total) out[idx] = f2bf(in[idx]);
}

// W[K,M] f32 -> Wt[M,K] bf16
__global__ __launch_bounds__(256)
void convT_bf16(const float* __restrict__ W, __bf16* __restrict__ Wt, int K, int M)
{
    int idx = blockIdx.x * blockDim.x + threadIdx.x;
    if (idx >= K * M) return;
    int k = idx / M, m = idx - k * M;
    Wt[(size_t)m * K + k] = f2bf(W[idx]);
}

// ---------------------------------------------------------------------------
// SDDMM: per (edge, head) score = q[dst,h,:] . k[src,h,:]  (bf16 gathers),
// plus running segment max via encoded atomicMax.
// ---------------------------------------------------------------------------
__global__ __launch_bounds__(256)
void sddmm_max(const __bf16* __restrict__ q, const __bf16* __restrict__ kmat,
               const int* __restrict__ src, const int* __restrict__ dst,
               float* __restrict__ score, unsigned* __restrict__ mx, int total)
{
    int idx = blockIdx.x * blockDim.x + threadIdx.x;
    if (idx >= total) return;
    int eid = idx >> 3, hh = idx & 7;
    int d = dst[eid], s = src[eid];
    const uint4* qp = (const uint4*)(q    + (size_t)d * HID_F + hh * DH);
    const uint4* kp = (const uint4*)(kmat + (size_t)s * HID_F + hh * DH);
    float acc = 0.f;
#pragma unroll
    for (int j = 0; j < 4; ++j) {
        uint4 a = qp[j], b = kp[j];
        unsigned au[4] = {a.x, a.y, a.z, a.w};
        unsigned bu[4] = {b.x, b.y, b.z, b.w};
#pragma unroll
        for (int c = 0; c < 4; ++c) {
            float2 fa = bfpair(au[c]), fb = bfpair(bu[c]);
            acc = fmaf(fa.x, fb.x, acc);
            acc = fmaf(fa.y, fb.y, acc);
        }
    }
    score[idx] = acc;
    atomicMax(&mx[(size_t)d * HEADS + hh], fenc(acc));
}

// exp(score - max[dst]) in place, accumulate segment sums
__global__ __launch_bounds__(256)
void exp_sum(const int* __restrict__ dst, const unsigned* __restrict__ mx,
             float* __restrict__ score, float* __restrict__ ssum, int total)
{
    int idx = blockIdx.x * blockDim.x + threadIdx.x;
    if (idx >= total) return;
    int eid = idx >> 3, hh = idx & 7;
    int d = dst[eid];
    float m = fdec(mx[(size_t)d * HEADS + hh]);
    float ex = __expf(score[idx] - m);
    score[idx] = ex;
    atomicAdd(&ssum[(size_t)d * HEADS + hh], ex);
}

// SpMM scatter: acc[dst] += w * v[src]   (bf16 gathers, f32 atomics)
__global__ __launch_bounds__(256)
void spmm_scatter(const float* __restrict__ attn, const __bf16* __restrict__ vmat,
                  const int* __restrict__ src, const int* __restrict__ dst,
                  float* __restrict__ acc, int total)
{
    int idx = blockIdx.x * blockDim.x + threadIdx.x;
    if (idx >= total) return;
    int eid = idx >> 3, hh = idx & 7;
    float w = attn[idx];
    int s = src[eid], d = dst[eid];
    const uint4* vp = (const uint4*)(vmat + (size_t)s * HID_F + hh * DH);
    float* op = acc + (size_t)d * HID_F + hh * DH;
#pragma unroll
    for (int j = 0; j < 4; ++j) {
        uint4 t = vp[j];
        unsigned tu[4] = {t.x, t.y, t.z, t.w};
#pragma unroll
        for (int c = 0; c < 4; ++c) {
            float2 f = bfpair(tu[c]);
            atomicAdd(op + j * 8 + c * 2 + 0, w * f.x);
            atomicAdd(op + j * 8 + c * 2 + 1, w * f.y);
        }
    }
}

// divide accumulated rows by softmax denominator, emit bf16 features
__global__ __launch_bounds__(256)
void norm_div_bf16(const float* __restrict__ ssum, const float* __restrict__ acc,
                   __bf16* __restrict__ hb, int total)
{
    int idx = blockIdx.x * blockDim.x + threadIdx.x;
    if (idx >= total) return;
    int nrow = idx >> 8;           // / HID_F
    int hh   = (idx & 255) >> 5;   // head
    float s = ssum[(size_t)nrow * HEADS + hh];
    float v = acc[idx];
    hb[idx] = f2bf((s > 0.f) ? v / s : 0.f);
}

// ---------------------------------------------------------------------------
extern "C" void kernel_launch(void* const* d_in, const int* in_sizes, int n_in,
                              void* d_out, int out_size, void* d_ws, size_t ws_size,
                              hipStream_t stream)
{
    (void)n_in; (void)out_size; (void)ws_size;

    const float* X    = (const float*)d_in[0];
    const int*   src  = (const int*)  d_in[1];
    const int*   dst  = (const int*)  d_in[2];
    const float* Win  = (const float*)d_in[3];
    const float* bin_ = (const float*)d_in[4];
    const float* Wq   = (const float*)d_in[5];
    const float* Wk   = (const float*)d_in[6];
    const float* Wv   = (const float*)d_in[7];
    const float* bq   = (const float*)d_in[8];
    const float* bk   = (const float*)d_in[9];
    const float* bv   = (const float*)d_in[10];
    const float* Wout = (const float*)d_in[11];
    const float* bout = (const float*)d_in[12];
    float* out = (float*)d_out;

    const int N = in_sizes[0] / IN_F;
    const int E = in_sizes[1];

    const size_t NH  = (size_t)N * HID_F;
    const size_t EH  = (size_t)E * HEADS;
    const size_t NHd = (size_t)N * HEADS;

    // ---- workspace layout (bf16 A-buffers placed with slack after them) ----
    char* p = (char*)d_ws;
    auto alloc = [&](size_t bytes) -> char* {
        char* r = p; p += (bytes + 255) & ~(size_t)255; return r;
    };
    __bf16* Xbf  = (__bf16*)alloc((size_t)N * IN_F * 2);
    __bf16* hb0  = (__bf16*)alloc(NH * 2);
    __bf16* hb1  = (__bf16*)alloc(NH * 2);
    __bf16* qb   = (__bf16*)alloc(NH * 2);
    __bf16* kb   = (__bf16*)alloc(NH * 2);
    __bf16* vb   = (__bf16*)alloc(NH * 2);
    __bf16* WtIn = (__bf16*)alloc((size_t)IN_F * HID_F * 2);
    __bf16* WtQ  = (__bf16*)alloc((size_t)NLAYER * HID_F * HID_F * 2);
    __bf16* WtK  = (__bf16*)alloc((size_t)NLAYER * HID_F * HID_F * 2);
    __bf16* WtV  = (__bf16*)alloc((size_t)NLAYER * HID_F * HID_F * 2);
    __bf16* WtO  = (__bf16*)alloc((size_t)HID_F * OUT_F * 2);
    float*    attn = (float*)   alloc(EH * 4);
    unsigned* mx   = (unsigned*)alloc(NHd * 4);
    float*    ssum = (float*)   alloc(NHd * 4);
    float*    accf = (float*)   alloc(NH * 4);

    const float scaling = 1.0f / sqrtf((float)DH);

    auto gemm = [&](const __bf16* A, const __bf16* Wt, const float* b, void* C,
                    int n, int K, int M, float scale, int obf) {
        dim3 grid((M + BN - 1) / BN, (n + BM - 1) / BM);
        gemm_wmma<<<grid, 256, 0, stream>>>(A, Wt, b, C, n, K, M, scale, obf);
    };

    // ---- one-time conversions (graph-captured; deterministic) ----
    {
        int tot = N * IN_F;
        conv_bf16<<<(tot + 255) / 256, 256, 0, stream>>>(X, Xbf, tot);
        int wt1 = IN_F * HID_F;
        convT_bf16<<<(wt1 + 255) / 256, 256, 0, stream>>>(Win, WtIn, IN_F, HID_F);
        int wt2 = HID_F * HID_F;
        for (int l = 0; l < NLAYER; ++l) {
            size_t off = (size_t)l * HID_F * HID_F;
            convT_bf16<<<(wt2 + 255) / 256, 256, 0, stream>>>(Wq + off, WtQ + off, HID_F, HID_F);
            convT_bf16<<<(wt2 + 255) / 256, 256, 0, stream>>>(Wk + off, WtK + off, HID_F, HID_F);
            convT_bf16<<<(wt2 + 255) / 256, 256, 0, stream>>>(Wv + off, WtV + off, HID_F, HID_F);
        }
        int wt3 = HID_F * OUT_F;
        convT_bf16<<<(wt3 + 255) / 256, 256, 0, stream>>>(Wout, WtO, HID_F, OUT_F);
    }

    // input projection: h = X @ Win + bin  (bf16 out)
    gemm(Xbf, WtIn, bin_, hb0, N, IN_F, HID_F, 1.0f, 1);

    __bf16* hcur = hb0;
    __bf16* hnxt = hb1;

    const int etot = (int)EH;
    const int eblk = (etot + 255) / 256;
    const int ntot = (int)NH;
    const int nblk = (ntot + 255) / 256;

    for (int l = 0; l < NLAYER; ++l) {
        size_t woff = (size_t)l * HID_F * HID_F;
        const float* bql = bq + (size_t)l * HID_F;
        const float* bkl = bk + (size_t)l * HID_F;
        const float* bvl = bv + (size_t)l * HID_F;

        gemm(hcur, WtQ + woff, bql, qb, N, HID_F, HID_F, scaling, 1);
        gemm(hcur, WtK + woff, bkl, kb, N, HID_F, HID_F, 1.0f,    1);
        gemm(hcur, WtV + woff, bvl, vb, N, HID_F, HID_F, 1.0f,    1);

        (void)hipMemsetAsync(mx,   0, NHd * sizeof(unsigned), stream);
        (void)hipMemsetAsync(ssum, 0, NHd * sizeof(float),    stream);
        (void)hipMemsetAsync(accf, 0, NH  * sizeof(float),    stream);

        sddmm_max    <<<eblk, 256, 0, stream>>>(qb, kb, src, dst, attn, mx, etot);
        exp_sum      <<<eblk, 256, 0, stream>>>(dst, mx, attn, ssum, etot);
        spmm_scatter <<<eblk, 256, 0, stream>>>(attn, vb, src, dst, accf, etot);
        norm_div_bf16<<<nblk, 256, 0, stream>>>(ssum, accf, hnxt, ntot);

        __bf16* t = hcur; hcur = hnxt; hnxt = t;
    }

    // predictor: out = h @ Wout + bout  (f32 out)
    gemm(hcur, WtO, bout, out, N, HID_F, OUT_F, 1.0f, 0);
}
